// TorchMultiheadAttention_46325517254850
// MI455X (gfx1250) — matrix operand order repaired and checked by reading, not compiled, not run on hardware
//
#include <hip/hip_runtime.h>
#include <stdint.h>

#define EMBED 512
#define HEADS 8
#define HDIM  64
#define BATCH 4
#define SEQ   2048
#define NTOK  (BATCH * SEQ)          // 8192 tokens

typedef __attribute__((ext_vector_type(16))) __bf16 v16bf;
typedef __attribute__((ext_vector_type(8)))  float  v8f;

union FragB16 {
  v16bf v;
  unsigned short s[16];
  uint4 q[2];
};

__device__ __forceinline__ unsigned short f32_to_bf16_bits(float f) {
  union { float f; uint32_t u; } cv; cv.f = f;
  uint32_t u = cv.u;
  u += 0x7FFFu + ((u >> 16) & 1u);   // round-to-nearest-even
  return (unsigned short)(u >> 16);
}

#define WMMA_BF16(A, B, C) \
  __builtin_amdgcn_wmma_f32_16x16x32_bf16(false, (A), false, (B), (short)0, (C), false, false)

// A-fragment (16x32 MxK) from bf16 row-major tile, row stride ld (elements).
// Lane l (r=l&15, h=l>>4): a[0..15]=A[r][16h+j].
__device__ __forceinline__ v16bf load_a_bf16(const unsigned short* tile, int ld, int lane) {
  const int r = lane & 15, h = lane >> 4;
  const unsigned short* p = tile + (size_t)r * ld + 16 * h;
  FragB16 f;
  f.q[0] = *(const uint4*)(p);
  f.q[1] = *(const uint4*)(p + 8);
  return f.v;
}

// B-fragment lane run: 16 contiguous bf16 at p (caller computed per-lane base;
// B[k][n] comes from row n of a K-major matrix, so the run is contiguous).
__device__ __forceinline__ v16bf load_b_bf16(const unsigned short* p) {
  FragB16 f;
  f.q[0] = *(const uint4*)(p);
  f.q[1] = *(const uint4*)(p + 8);
  return f.v;
}

// ---------------------------------------------------------------------------
// Kernel 0: one-shot fp32 -> bf16 conversion (packed float4 -> 4 bf16).
// Pays the RNE conversion VALU exactly once per element, off the GEMM path.
// ---------------------------------------------------------------------------
__global__ __launch_bounds__(256) void cvt_bf16_kernel(
    const float* __restrict__ in, unsigned short* __restrict__ out, int n4) {
  const int i = blockIdx.x * blockDim.x + threadIdx.x;
  if (i >= n4) return;
  const float4 f = ((const float4*)in)[i];
  ushort4 o;
  o.x = f32_to_bf16_bits(f.x);
  o.y = f32_to_bf16_bits(f.y);
  o.z = f32_to_bf16_bits(f.z);
  o.w = f32_to_bf16_bits(f.w);
  ((ushort4*)out)[i] = o;
}

// ---------------------------------------------------------------------------
// Shared 32(M) x 64(N) x 512(K) bf16 GEMM wave body: 8 WMMAs per K-step,
// pure b128 loads in the loop (zero conversion VALU).
// A: bf16 row-major [M x 512]; B: bf16 row-major weight [N x 512] (used as B^T).
// ---------------------------------------------------------------------------
__device__ __forceinline__ void gemm_wave_32x64(
    const unsigned short* __restrict__ A,  // &A[m0][0]
    const unsigned short* __restrict__ B,  // &W[n0][0]
    v8f acc[2][4], int lane) {
  const int r = lane & 15, h = lane >> 4;
  for (int kb = 0; kb < EMBED; kb += 32) {
    __builtin_prefetch(A + (size_t)r * EMBED + kb + 64, 0, 0);
    __builtin_prefetch(B + (size_t)r * EMBED + kb + 64, 0, 0);
    v16bf a0 = load_a_bf16(A + kb, EMBED, lane);
    v16bf a1 = load_a_bf16(A + (size_t)16 * EMBED + kb, EMBED, lane);
#pragma unroll
    for (int t = 0; t < 4; ++t) {
      const unsigned short* bp = B + (size_t)(t * 16 + r) * EMBED + kb + 16 * h;
      v16bf b = load_b_bf16(bp);
      acc[0][t] = WMMA_BF16(a0, b, acc[0][t]);
      acc[1][t] = WMMA_BF16(a1, b, acc[1][t]);
    }
  }
}

// ---------------------------------------------------------------------------
// Kernel 1: QKV projection.  out = X_z @ W_z^T + b_z, z in {q,k,v}.
// Q,K stored bf16 [B,H,S,D]; V stored bf16 transposed [B,H,D,S].
// Block = 256 thr = 8 waves; each wave computes a 32x64 output tile.
// ---------------------------------------------------------------------------
__global__ __launch_bounds__(256) void qkv_proj_kernel(
    const unsigned short* __restrict__ xq, const unsigned short* __restrict__ xk,
    const unsigned short* __restrict__ xv, const unsigned short* __restrict__ wi,
    const float* __restrict__ bias,
    unsigned short* __restrict__ q_ws, unsigned short* __restrict__ k_ws,
    unsigned short* __restrict__ vt_ws) {
  const int z = blockIdx.z;
  const unsigned short* x  = (z == 0) ? xq : (z == 1) ? xk : xv;
  const unsigned short* wz = wi + (size_t)z * EMBED * EMBED;
  const float* bz = bias + z * EMBED;
  const int wave = threadIdx.x >> 5, lane = threadIdx.x & 31;
  const int r = lane & 15, hh = lane >> 4;
  const int m0 = (blockIdx.x * 8 + wave) * 32;
  const int n0 = blockIdx.y * 64;

  v8f acc[2][4] = {};
  gemm_wave_32x64(x + (size_t)m0 * EMBED, wz + (size_t)n0 * EMBED, acc, lane);

#pragma unroll
  for (int t = 0; t < 4; ++t) {
    const int n = n0 + t * 16 + r;       // output channel
    const int hd = n >> 6, dd = n & 63;  // head, dim-in-head
    const float bv = bz[n];
#pragma unroll
    for (int sub = 0; sub < 2; ++sub) {
#pragma unroll
      for (int v = 0; v < 8; ++v) {
        const int m = m0 + sub * 16 + v + 8 * hh;  // token index
        const int b = m >> 11, s = m & (SEQ - 1);
        const unsigned short val = f32_to_bf16_bits(acc[sub][t][v] + bv);
        if (z == 0)
          q_ws[((size_t)(b * HEADS + hd) * SEQ + s) * HDIM + dd] = val;
        else if (z == 1)
          k_ws[((size_t)(b * HEADS + hd) * SEQ + s) * HDIM + dd] = val;
        else
          vt_ws[((size_t)(b * HEADS + hd) * HDIM + dd) * SEQ + s] = val;
      }
    }
  }
}

// ---------------------------------------------------------------------------
// Kernel 2: flash-style attention per (b, h).
// Block = 128 thr = 4 waves; wave owns 16 q-rows, streams keys in 32-chunks
// with online softmax (exp2 domain).  Output: attended bf16 [B,S,E].
// ---------------------------------------------------------------------------
__global__ __launch_bounds__(128) void attn_kernel(
    const unsigned short* __restrict__ q_ws,
    const unsigned short* __restrict__ k_ws,
    const unsigned short* __restrict__ vt_ws,
    unsigned short* __restrict__ att_ws) {
  __shared__ __align__(16) unsigned short pbuf[4][16 * 32];  // per-wave P tile
  const int wave = threadIdx.x >> 5, lane = threadIdx.x & 31;
  const int r = lane & 15, hh = lane >> 4;
  const int b = blockIdx.z, head = blockIdx.y;
  const int q0 = blockIdx.x * 64 + wave * 16;
  const size_t bh = (size_t)(b * HEADS + head);
  const unsigned short* Q  = q_ws  + bh * SEQ * HDIM;
  const unsigned short* K  = k_ws  + bh * SEQ * HDIM;
  const unsigned short* VT = vt_ws + bh * HDIM * SEQ;

  // Preload Q fragments for d = 0..31 and 32..63.
  v16bf qa0 = load_a_bf16(Q + (size_t)q0 * HDIM + 0,  HDIM, lane);
  v16bf qa1 = load_a_bf16(Q + (size_t)q0 * HDIM + 32, HDIM, lane);

  const float scale = 0.125f * 1.44269504f;  // 1/sqrt(64) * log2(e)
  float mrun[8], lrun[8];
  v8f o[4] = {};
#pragma unroll
  for (int v = 0; v < 8; ++v) { mrun[v] = -1e30f; lrun[v] = 0.0f; }

  for (int kb = 0; kb < SEQ; kb += 32) {
    __builtin_prefetch(K + (size_t)(kb + 32 + r) * HDIM, 0, 0);
    // --- scores: two 16q x 16key tiles, K-dim = 64 in two WMMA steps ---
    v8f c0 = {}, c1 = {};
    {
      const unsigned short* kA = K + (size_t)(kb + r) * HDIM + 16 * hh;
      const unsigned short* kB = K + (size_t)(kb + 16 + r) * HDIM + 16 * hh;
      c0 = WMMA_BF16(qa0, load_b_bf16(kA),      c0);
      c0 = WMMA_BF16(qa1, load_b_bf16(kA + 32), c0);
      c1 = WMMA_BF16(qa0, load_b_bf16(kB),      c1);
      c1 = WMMA_BF16(qa1, load_b_bf16(kB + 32), c1);
    }
    // --- online softmax update (row = (v, half); 16-lane group reduction) ---
    float alpha[8], p0[8], p1[8];
#pragma unroll
    for (int v = 0; v < 8; ++v) {
      const float s0 = c0[v] * scale, s1 = c1[v] * scale;
      float t = fmaxf(s0, s1);
      t = fmaxf(t, __shfl_xor(t, 1, 32));
      t = fmaxf(t, __shfl_xor(t, 2, 32));
      t = fmaxf(t, __shfl_xor(t, 4, 32));
      t = fmaxf(t, __shfl_xor(t, 8, 32));
      const float mnew = fmaxf(mrun[v], t);
      alpha[v] = exp2f(mrun[v] - mnew);
      p0[v] = exp2f(s0 - mnew);
      p1[v] = exp2f(s1 - mnew);
      float rs = p0[v] + p1[v];
      rs += __shfl_xor(rs, 1, 32);
      rs += __shfl_xor(rs, 2, 32);
      rs += __shfl_xor(rs, 4, 32);
      rs += __shfl_xor(rs, 8, 32);
      lrun[v] = lrun[v] * alpha[v] + rs;
      mrun[v] = mnew;
    }
#pragma unroll
    for (int v = 0; v < 8; ++v) {
      o[0][v] *= alpha[v]; o[1][v] *= alpha[v];
      o[2][v] *= alpha[v]; o[3][v] *= alpha[v];
    }
    // --- C-layout -> A-layout through per-wave LDS tile (16 x 32 bf16) ---
    unsigned short* pb = pbuf[wave];
#pragma unroll
    for (int v = 0; v < 8; ++v) {
      const int row = v + 8 * hh;
      pb[row * 32 + r]      = f32_to_bf16_bits(p0[v]);
      pb[row * 32 + 16 + r] = f32_to_bf16_bits(p1[v]);
    }
    FragB16 fp;  // A-fragment of P from LDS (same wave; DScnt keeps order)
    {
      const unsigned short* p = pb + r * 32 + 16 * hh;
      fp.q[0] = *(const uint4*)(p);
      fp.q[1] = *(const uint4*)(p + 8);
    }
    // --- PV: 4 d-tiles, K-dim = 32 keys, B from transposed V ---
#pragma unroll
    for (int t = 0; t < 4; ++t) {
      const unsigned short* vp = VT + (size_t)(t * 16 + r) * SEQ + kb + 16 * hh;
      o[t] = WMMA_BF16(fp.v, load_b_bf16(vp), o[t]);
    }
  }

  // --- normalize and write attended bf16 [B,S,E] (token-major for out-proj) ---
#pragma unroll
  for (int v = 0; v < 8; ++v) {
    const float inv = 1.0f / lrun[v];
    const int m = q0 + v + 8 * hh;
#pragma unroll
    for (int t = 0; t < 4; ++t) {
      att_ws[((size_t)(b * SEQ + m)) * EMBED + head * HDIM + t * 16 + r] =
          f32_to_bf16_bits(o[t][v] * inv);
    }
  }
}

// ---------------------------------------------------------------------------
// Kernel 3: output projection.  out = att @ Wout^T + bias  (fp32 out).
// ---------------------------------------------------------------------------
__global__ __launch_bounds__(256) void out_proj_kernel(
    const unsigned short* __restrict__ att, const unsigned short* __restrict__ wo,
    const float* __restrict__ bias, float* __restrict__ out) {
  const int wave = threadIdx.x >> 5, lane = threadIdx.x & 31;
  const int r = lane & 15, hh = lane >> 4;
  const int m0 = (blockIdx.x * 8 + wave) * 32;
  const int n0 = blockIdx.y * 64;

  v8f acc[2][4] = {};
  gemm_wave_32x64(att + (size_t)m0 * EMBED, wo + (size_t)n0 * EMBED, acc, lane);

#pragma unroll
  for (int t = 0; t < 4; ++t) {
    const int n = n0 + t * 16 + r;
    const float bv = bias[n];
#pragma unroll
    for (int sub = 0; sub < 2; ++sub) {
#pragma unroll
      for (int v = 0; v < 8; ++v) {
        const int m = m0 + sub * 16 + v + 8 * hh;
        out[(size_t)m * EMBED + n] = acc[sub][t][v] + bv;
      }
    }
  }
}

// ---------------------------------------------------------------------------
extern "C" void kernel_launch(void* const* d_in, const int* in_sizes, int n_in,
                              void* d_out, int out_size, void* d_ws, size_t ws_size,
                              hipStream_t stream) {
  const float* q_in  = (const float*)d_in[0];
  const float* k_in  = (const float*)d_in[1];
  const float* v_in  = (const float*)d_in[2];
  const float* in_w  = (const float*)d_in[3];
  const float* in_b  = (const float*)d_in[4];
  const float* out_w = (const float*)d_in[5];
  const float* out_b = (const float*)d_in[6];

  const size_t act   = (size_t)NTOK * EMBED;       // 4,194,304 elements
  const size_t wi_n  = (size_t)3 * EMBED * EMBED;  //   786,432
  const size_t wo_n  = (size_t)EMBED * EMBED;      //   262,144

  unsigned short* p = (unsigned short*)d_ws;
  unsigned short* xq_bf  = p; p += act;
  unsigned short* xk_bf  = p; p += act;
  unsigned short* xv_bf  = p; p += act;
  unsigned short* wi_bf  = p; p += wi_n;
  unsigned short* wo_bf  = p; p += wo_n;
  unsigned short* q_ws   = p; p += act;
  unsigned short* k_ws   = p; p += act;
  unsigned short* vt_ws  = p; p += act;
  unsigned short* att_ws = xq_bf;  // alias: xq dead after qkv_proj completes

  // 0) One-shot fp32 -> bf16 conversions (packed, memory bound).
  {
    const int tb = 256;
    cvt_bf16_kernel<<<dim3((int)(act / 4 + tb - 1) / tb), tb, 0, stream>>>(q_in, xq_bf, (int)(act / 4));
    cvt_bf16_kernel<<<dim3((int)(act / 4 + tb - 1) / tb), tb, 0, stream>>>(k_in, xk_bf, (int)(act / 4));
    cvt_bf16_kernel<<<dim3((int)(act / 4 + tb - 1) / tb), tb, 0, stream>>>(v_in, xv_bf, (int)(act / 4));
    cvt_bf16_kernel<<<dim3((int)(wi_n / 4 + tb - 1) / tb), tb, 0, stream>>>(in_w, wi_bf, (int)(wi_n / 4));
    cvt_bf16_kernel<<<dim3((int)(wo_n / 4 + tb - 1) / tb), tb, 0, stream>>>(out_w, wo_bf, (int)(wo_n / 4));
  }

  // 1) QKV projections (z = 0/1/2 selects q/k/v input + weight slice).
  qkv_proj_kernel<<<dim3(NTOK / 256, EMBED / 64, 3), 256, 0, stream>>>(
      xq_bf, xk_bf, xv_bf, wi_bf, in_b, q_ws, k_ws, vt_ws);

  // 2) Flash attention: (q-block of 64, head, batch).
  attn_kernel<<<dim3(SEQ / 64, HEADS, BATCH), 128, 0, stream>>>(
      q_ws, k_ws, vt_ws, att_ws);

  // 3) Output projection into fp32 d_out.
  out_proj_kernel<<<dim3(NTOK / 256, EMBED / 64), 256, 0, stream>>>(
      att_ws, out_w ? wo_bf : wo_bf, out_b, (float*)d_out);
}
